// GNN_prelu_50689204027576
// MI455X (gfx1250) — matrix operand order repaired and checked by reading, not compiled, not run on hardware
//
#include <hip/hip_runtime.h>
#include <hip/hip_bf16.h>

// ---------------------------------------------------------------------------
// Problem constants (from the reference)
// ---------------------------------------------------------------------------
constexpr int N_PFAS = 50000, N_GW = 100000, N_SW = 50000;
constexpr int D = 128;
constexpr int E_PG = 1000000, E_GP = 1000000, E_PS = 500000, E_SP = 500000;
constexpr int E_TOTAL = E_PG + E_PS + E_GP + E_SP;     // 3,000,000 (order: pg, ps, gp, sp)
constexpr int WAVES_SC = E_TOTAL / 32;                 // 93,750 (all boundaries % 32 == 0)

typedef __attribute__((ext_vector_type(16))) __bf16 v16bf;
typedef __attribute__((ext_vector_type(8)))  float  v8f;

// ---------------------------------------------------------------------------
// Convert the 8 weight matrices (128x128 fp32, row = out-channel n, col = k)
// to bf16 copies in workspace. 8 * 16384 elements.
// ---------------------------------------------------------------------------
__global__ __launch_bounds__(256) void cvt_w_kernel(
    const float* __restrict__ s0, const float* __restrict__ s1,
    const float* __restrict__ s2, const float* __restrict__ s3,
    const float* __restrict__ s4, const float* __restrict__ s5,
    const float* __restrict__ s6, const float* __restrict__ s7,
    __bf16* __restrict__ out) {
  int i = blockIdx.x * blockDim.x + threadIdx.x;   // 0 .. 131071
  const float* srcs[8] = {s0, s1, s2, s3, s4, s5, s6, s7};
  out[i] = (__bf16)srcs[i >> 14][i & 16383];
}

// ---------------------------------------------------------------------------
// Merged edge scatter over all 4 relations (one launch -> full occupancy for
// the whole atomic phase).  Each wave owns a 32-edge chunk: lanes load the 32
// src/dst indices coalesced, per-edge counts are bumped one-atomic-per-lane,
// then the wave iterates its 32 edges with v_readlane broadcasts so the
// feature gather (float4/lane) and the 4 global_atomic_add_f32 per lane use
// scalar-base addressing.  Accumulators (128 MB) are L2-resident (192 MB L2).
// ---------------------------------------------------------------------------
__global__ __launch_bounds__(256) void scatter_all_kernel(
    const float* __restrict__ xp, const float* __restrict__ xg,
    const float* __restrict__ xs,
    const int* __restrict__ pg_s, const int* __restrict__ pg_d,
    const int* __restrict__ ps_s, const int* __restrict__ ps_d,
    const int* __restrict__ gp_s, const int* __restrict__ gp_d,
    const int* __restrict__ sp_s, const int* __restrict__ sp_d,
    float* agg_gw, float* cnt_gw, float* agg_sw, float* cnt_sw,
    float* agg_gp, float* cnt_gp, float* agg_sp, float* cnt_sp) {
  const int lane = threadIdx.x & 31;
  const int wave = blockIdx.x * 8 + (threadIdx.x >> 5);
  if (wave >= WAVES_SC) return;
  int base = wave * 32;

  const float* xsrc; const int* srcp; const int* dstp; float* agg; float* cnt;
  if (base < E_PG) {
    xsrc = xp; srcp = pg_s; dstp = pg_d; agg = agg_gw; cnt = cnt_gw;
  } else if (base < E_PG + E_PS) {
    base -= E_PG;               xsrc = xp; srcp = ps_s; dstp = ps_d; agg = agg_sw; cnt = cnt_sw;
  } else if (base < E_PG + E_PS + E_GP) {
    base -= E_PG + E_PS;        xsrc = xg; srcp = gp_s; dstp = gp_d; agg = agg_gp; cnt = cnt_gp;
  } else {
    base -= E_PG + E_PS + E_GP; xsrc = xs; srcp = sp_s; dstp = sp_d; agg = agg_sp; cnt = cnt_sp;
  }

  const int sv = srcp[base + lane];     // coalesced: 32 edges' src indices
  const int dv = dstp[base + lane];     // coalesced: 32 edges' dst indices
  unsafeAtomicAdd(cnt + dv, 1.0f);      // per-edge count, one atomic per lane

#pragma unroll 4
  for (int i = 0; i < 32; ++i) {
    const int s = __builtin_amdgcn_readlane(sv, i);   // uniform -> SGPR base
    const int d = __builtin_amdgcn_readlane(dv, i);
    const float4 v = ((const float4*)(xsrc + (size_t)s * D))[lane];
    float* ag = agg + (size_t)d * D + lane * 4;
    unsafeAtomicAdd(ag + 0, v.x);
    unsafeAtomicAdd(ag + 1, v.y);
    unsafeAtomicAdd(ag + 2, v.z);
    unsafeAtomicAdd(ag + 3, v.w);
  }
}

// ---------------------------------------------------------------------------
// Fused SAGE GEMM:  out = relu( sum_seg (scale_seg * A_seg) @ W_seg^T + bias )
//   - segment with cnt != nullptr: A row scaled by 1/max(cnt,1)  (mean agg)
//   - segment with cnt == nullptr: root/self features, scale 1
// Block = 256 threads = 8 waves; wave w owns rows [blk*128 + w*16, +16) and
// all 8 n-tiles (v8f acc[8]).  v_wmma_f32_16x16x32_bf16, fp32 accumulate.
// All 8 B fragments of a k-step are preloaded so the global_load_b128 clause
// drains across the WMMA chain instead of a wait before every WMMA.
// No LDS / no barriers -> partial waves exit before any WMMA (EXEC all-ones).
// ---------------------------------------------------------------------------
struct SegDesc {
  const float*  A;    // [M,128] fp32
  const float*  cnt;  // [M] or nullptr
  const __bf16* W;    // [128,128] bf16, row n, col k
};

template <int NSEG>
__global__ __launch_bounds__(256) void sage_gemm_kernel(
    SegDesc s0, SegDesc s1, SegDesc s2, SegDesc s3,
    const float* __restrict__ bias0, const float* __restrict__ bias1,
    float* __restrict__ out, int M) {
  const int lane  = threadIdx.x & 31;
  const int wid   = threadIdx.x >> 5;
  const int m0    = blockIdx.x * 128 + wid * 16;
  if (m0 >= M) return;                  // whole-wave exit (M % 16 == 0)
  const int rowL  = lane & 15;          // A row / B col / C col within tile
  const int khalf = lane >> 4;          // which K half this lane holds
  const int row   = m0 + rowL;

  v8f acc[8] = {};                      // 8 n-tiles covering N = 128

  const SegDesc segs[4] = {s0, s1, s2, s3};
#pragma unroll
  for (int sg = 0; sg < NSEG; ++sg) {
    const float*  A = segs[sg].A;
    const __bf16* W = segs[sg].W;
    float scale = 1.0f;
    if (segs[sg].cnt) scale = 1.0f / fmaxf(segs[sg].cnt[row], 1.0f);
    const float* arow = A + (size_t)row * D;
#pragma unroll
    for (int ks = 0; ks < 4; ++ks) {    // K = 128 per segment, 32 per WMMA
      const int kb = ks * 32;
      // ---- A fragment: lane holds row rowL, K = kb + 16*khalf .. +15
      float fv[16];
      {
        const float4 f0 = *(const float4*)(arow + kb + 16 * khalf);
        const float4 f1 = *(const float4*)(arow + kb + 16 * khalf + 4);
        const float4 f2 = *(const float4*)(arow + kb + 16 * khalf + 8);
        const float4 f3 = *(const float4*)(arow + kb + 16 * khalf + 12);
        fv[0] = f0.x; fv[1] = f0.y; fv[2]  = f0.z; fv[3]  = f0.w;
        fv[4] = f1.x; fv[5] = f1.y; fv[6]  = f1.z; fv[7]  = f1.w;
        fv[8] = f2.x; fv[9] = f2.y; fv[10] = f2.z; fv[11] = f2.w;
        fv[12]= f3.x; fv[13]= f3.y; fv[14] = f3.z; fv[15] = f3.w;
      }
      v16bf a;
#pragma unroll
      for (int i = 0; i < 16; ++i) a[i] = (__bf16)(fv[i] * scale);
      // ---- preload all 8 B fragments (16 contiguous bf16 per lane each)
      v16bf bfr[8];
#pragma unroll
      for (int nt = 0; nt < 8; ++nt) {
        const int n = nt * 16 + rowL;
        bfr[nt] = *(const v16bf*)(W + (size_t)n * D + kb + 16 * khalf);
      }
#pragma unroll
      for (int nt = 0; nt < 8; ++nt) {
        acc[nt] = __builtin_amdgcn_wmma_f32_16x16x32_bf16(
            false, a, false, bfr[nt], (short)0, acc[nt], false, false);
      }
    }
  }

  // ---- epilogue: bias + ReLU, coalesced stores (col = lane%16 contiguous)
#pragma unroll
  for (int nt = 0; nt < 8; ++nt) {
    const int col = nt * 16 + rowL;
    float b = 0.0f;
    if (bias0) b += bias0[col];
    if (bias1) b += bias1[col];
#pragma unroll
    for (int r = 0; r < 8; ++r) {
      const int orow = m0 + r + 8 * khalf;
      float v = acc[nt][r] + b;
      v = fmaxf(v, 0.0f);
      out[(size_t)orow * D + col] = v;
    }
  }
}

// ---------------------------------------------------------------------------
// Head: out[i] = prelu( dot(h[i,:], w) + b, a ).  One wave per row, float4
// loads, wave32 xor-shuffle reduction.
// ---------------------------------------------------------------------------
__global__ __launch_bounds__(256) void head_kernel(
    const float* __restrict__ h, const float* __restrict__ w,
    const float* __restrict__ bptr, const float* __restrict__ aptr,
    float* __restrict__ out, int N) {
  const int lane = threadIdx.x & 31;
  const int rowi = (blockIdx.x * blockDim.x + threadIdx.x) >> 5;
  if (rowi >= N) return;
  const float4 hv = ((const float4*)(h + (size_t)rowi * D))[lane];
  const float4 wv = ((const float4*)w)[lane];
  float v = hv.x * wv.x + hv.y * wv.y + hv.z * wv.z + hv.w * wv.w;
#pragma unroll
  for (int off = 16; off > 0; off >>= 1) v += __shfl_xor(v, off, 32);
  if (lane == 0) {
    v += bptr[0];
    const float aw = aptr[0];
    out[rowi] = (v >= 0.0f) ? v : aw * v;
  }
}

// ---------------------------------------------------------------------------
// Launcher
// ---------------------------------------------------------------------------
extern "C" void kernel_launch(void* const* d_in, const int* in_sizes, int n_in,
                              void* d_out, int out_size, void* d_ws, size_t ws_size,
                              hipStream_t stream) {
  const float* x_pfas = (const float*)d_in[0];
  const float* x_gw   = (const float*)d_in[1];
  const float* x_sw   = (const float*)d_in[2];
  const int* ei_pg_src = (const int*)d_in[3];
  const int* ei_pg_dst = (const int*)d_in[4];
  const int* ei_gp_src = (const int*)d_in[5];
  const int* ei_gp_dst = (const int*)d_in[6];
  const int* ei_ps_src = (const int*)d_in[7];
  const int* ei_ps_dst = (const int*)d_in[8];
  const int* ei_sp_src = (const int*)d_in[9];
  const int* ei_sp_dst = (const int*)d_in[10];
  const float* Wl_pg = (const float*)d_in[11];
  const float* bl_pg = (const float*)d_in[12];
  const float* Wr_pg = (const float*)d_in[13];
  const float* Wl_gp = (const float*)d_in[14];
  const float* bl_gp = (const float*)d_in[15];
  const float* Wr_gp = (const float*)d_in[16];
  const float* Wl_ps = (const float*)d_in[17];
  const float* bl_ps = (const float*)d_in[18];
  const float* Wr_ps = (const float*)d_in[19];
  const float* Wl_sp = (const float*)d_in[20];
  const float* bl_sp = (const float*)d_in[21];
  const float* Wr_sp = (const float*)d_in[22];
  const float* W_gw = (const float*)d_in[23];
  const float* b_gw = (const float*)d_in[24];
  const float* W_sw = (const float*)d_in[25];
  const float* b_sw = (const float*)d_in[26];
  const float* a_gw = (const float*)d_in[27];
  const float* a_sw = (const float*)d_in[28];

  // ---- workspace carve (~129.3 MB) --------------------------------------
  float* agg_gw = (float*)d_ws;                 // [N_GW ,128] -> reused as h_gw
  float* agg_sw = agg_gw + (size_t)N_GW * D;    // [N_SW ,128] -> reused as h_sw
  float* agg_gp = agg_sw + (size_t)N_SW * D;    // [N_PFAS,128]
  float* agg_sp = agg_gp + (size_t)N_PFAS * D;  // [N_PFAS,128]
  float* cnt_gw = agg_sp + (size_t)N_PFAS * D;  // [N_GW]
  float* cnt_sw = cnt_gw + N_GW;                // [N_SW]
  float* cnt_gp = cnt_sw + N_SW;                // [N_PFAS]
  float* cnt_sp = cnt_gp + N_PFAS;              // [N_PFAS]
  __bf16* wbf   = (__bf16*)(cnt_sp + N_PFAS);   // 8 x [128,128] bf16 (32B aligned)

  const size_t zero_bytes =
      ((size_t)(N_GW + N_SW + 2 * N_PFAS) * D + (N_GW + N_SW + 2 * N_PFAS)) * 4;
  hipMemsetAsync(d_ws, 0, zero_bytes, stream);

  // bf16 weight copies: order Wl_pg, Wr_pg, Wl_ps, Wr_ps, Wl_gp, Wr_gp, Wl_sp, Wr_sp
  cvt_w_kernel<<<512, 256, 0, stream>>>(Wl_pg, Wr_pg, Wl_ps, Wr_ps,
                                        Wl_gp, Wr_gp, Wl_sp, Wr_sp, wbf);

  // ---- merged edge scatter (mean-agg numerators + counts) ---------------
  scatter_all_kernel<<<(WAVES_SC + 7) / 8, 256, 0, stream>>>(
      x_pfas, x_gw, x_sw,
      ei_pg_src, ei_pg_dst, ei_ps_src, ei_ps_dst,
      ei_gp_src, ei_gp_dst, ei_sp_src, ei_sp_dst,
      agg_gw, cnt_gw, agg_sw, cnt_sw, agg_gp, cnt_gp, agg_sp, cnt_sp);

  // ---- fused mean + GEMM + bias + ReLU (WMMA) ---------------------------
  float* h_pf   = (float*)d_out;                       // [N_PFAS,128]
  float* out_gw = h_pf + (size_t)N_PFAS * D;           // [N_GW]
  float* out_sw = out_gw + N_GW;                       // [N_SW]

  SegDesc gw0{agg_gw, cnt_gw, wbf + 0 * 16384};
  SegDesc gw1{x_gw,  nullptr, wbf + 1 * 16384};
  sage_gemm_kernel<2><<<(N_GW + 127) / 128, 256, 0, stream>>>(
      gw0, gw1, gw0, gw0, bl_pg, nullptr, /*h_gw in-place*/ agg_gw, N_GW);

  SegDesc sw0{agg_sw, cnt_sw, wbf + 2 * 16384};
  SegDesc sw1{x_sw,  nullptr, wbf + 3 * 16384};
  sage_gemm_kernel<2><<<(N_SW + 127) / 128, 256, 0, stream>>>(
      sw0, sw1, sw0, sw0, bl_ps, nullptr, /*h_sw in-place*/ agg_sw, N_SW);

  SegDesc p0{agg_gp, cnt_gp, wbf + 4 * 16384};
  SegDesc p1{x_pfas, nullptr, wbf + 5 * 16384};
  SegDesc p2{agg_sp, cnt_sp, wbf + 6 * 16384};
  SegDesc p3{x_pfas, nullptr, wbf + 7 * 16384};
  sage_gemm_kernel<4><<<(N_PFAS + 127) / 128, 256, 0, stream>>>(
      p0, p1, p2, p3, bl_gp, bl_sp, h_pf, N_PFAS);

  // ---- prediction heads -------------------------------------------------
  head_kernel<<<(N_GW * 32) / 256, 256, 0, stream>>>(agg_gw, W_gw, b_gw, a_gw, out_gw, N_GW);
  head_kernel<<<(N_SW * 32) / 256, 256, 0, stream>>>(agg_sw, W_sw, b_sw, a_sw, out_sw, N_SW);
}
